// LMU_37795712205130
// MI455X (gfx1250) — matrix operand order repaired and checked by reading, not compiled, not run on hardware
//
#include <hip/hip_runtime.h>
#include <hip/hip_bf16.h>

// Problem dims (fixed by the reference).
#define B_BATCH 16384
#define K_X     128
#define N_H     512
#define D_M     256
#define K_TOT   (K_X + N_H + D_M)   // 896; every 32-chunk falls in one segment

typedef __attribute__((ext_vector_type(16))) __bf16 v16bf;
typedef __attribute__((ext_vector_type(8)))  float  v8f;

#define LDSS 40                     // LDS row stride (bf16 elems): 32 + 8 pad, 80B rows (16B aligned)

// ---------------------------------------------------------------- bf16 helpers
__device__ __forceinline__ unsigned rnd_bf(float f) {     // RNE-rounded bits, bf16 in [31:16]
  union { float f; unsigned u; } v; v.f = f;
  return v.u + 0x7FFFu + ((v.u >> 16) & 1u);
}
__device__ __forceinline__ unsigned pack2(float a, float b) {   // {bf(b), bf(a)}
  return (rnd_bf(a) >> 16) | (rnd_bf(b) & 0xFFFF0000u);
}
__device__ __forceinline__ __bf16 f2bf(float f) {
  unsigned short hs = (unsigned short)(rnd_bf(f) >> 16);
  __bf16 o; __builtin_memcpy(&o, &hs, sizeof(o)); return o;
}

// ---------------------------------------------------------------- async copy (gfx1250)
// One instruction moves 16B/lane global -> LDS, tracked by ASYNCcnt.
#define ASYNC_LD_B128(ldsoff, gaddr)                                  \
  asm volatile("global_load_async_to_lds_b128 %0, %1, off"            \
               :: "v"(ldsoff), "v"(gaddr) : "memory")
#define WAIT_ASYNC(n) asm volatile("s_wait_asynccnt " #n ::: "memory")

__device__ __forceinline__ unsigned lds_off32(const void* p) {
  return (unsigned)(size_t)p;        // low 32 bits of LDS flat address = LDS offset
}
__device__ __forceinline__ unsigned long long gaddr64(const void* p) {
  return (unsigned long long)(size_t)p;
}

// ---------------------------------------------------------------- WMMA fragments (ISA 7.12.2)
// A: 16x32 bf16. lane = half*16 + row; elems 0..15 -> K = half*16+0..15
// (same K order as frag_b). Two 16B contiguous LDS reads.
__device__ __forceinline__ v16bf frag_a(const __bf16* As, int row16, int lane) {
  const int row = lane & 15, half = lane >> 4;
  const __bf16* p = As + (row16 + row) * LDSS;
  v16bf a;
#pragma unroll
  for (int e = 0; e < 16; ++e) a[e] = p[half * 16 + e];
  return a;
}
// B: 32x16 bf16. lanes 0-15 hold K=0..15, lanes 16-31 K=16..31; N = lane&15.
// LDS tile holds W[n][k] (row-major W) so this read IS the transpose.
__device__ __forceinline__ v16bf frag_b(const __bf16* Ws, int col16, int lane) {
  const int col = lane & 15, half = lane >> 4;
  const __bf16* p = Ws + (col16 + col) * LDSS;
  v16bf b;
#pragma unroll
  for (int e = 0; e < 16; ++e) b[e] = p[half * 16 + e];
  return b;
}

// ---------------------------------------------------------------- prep: u + bf16 activations
// One wave32 per batch row; fuses the u = x.ex + h.eh + m.em reduction with
// writing bf16 copies of x/h/m (the GEMM A-operands) -- conversion hoisted out
// of the GEMM hot loop, paid for by a pass we need anyway.
template <int CNT>  // CNT = columns/64 pairs-per-lane iterations
__device__ __forceinline__ float row_cvt_dot(const float* __restrict__ src,
                                             const float* __restrict__ e,
                                             unsigned* __restrict__ dst,
                                             int lane, float s) {
  const float2* s2 = (const float2*)src;
  const float2* e2 = (const float2*)e;
#pragma unroll
  for (int i = 0; i < CNT; ++i) {
    const int p = lane + 32 * i;
    const float2 v = s2[p], w = e2[p];
    s = fmaf(v.x, w.x, s);
    s = fmaf(v.y, w.y, s);
    dst[p] = pack2(v.x, v.y);
  }
  return s;
}

__global__ void lmu_prep_kernel(const float* __restrict__ x, const float* __restrict__ h,
                                const float* __restrict__ m, const float* __restrict__ ex,
                                const float* __restrict__ eh, const float* __restrict__ em,
                                __bf16* __restrict__ xb, __bf16* __restrict__ hb,
                                __bf16* __restrict__ mb, float* __restrict__ u) {
  const int lane = threadIdx.x & 31;
  const int row  = (int)((blockIdx.x * blockDim.x + threadIdx.x) >> 5);
  if (row >= B_BATCH) return;
  float s = 0.f;
  s = row_cvt_dot<K_X / 64>(x + (size_t)row * K_X, ex, (unsigned*)(xb + (size_t)row * K_X), lane, s);
  s = row_cvt_dot<N_H / 64>(h + (size_t)row * N_H, eh, (unsigned*)(hb + (size_t)row * N_H), lane, s);
  s = row_cvt_dot<D_M / 64>(m + (size_t)row * D_M, em, (unsigned*)(mb + (size_t)row * D_M), lane, s);
#pragma unroll
  for (int off = 16; off > 0; off >>= 1) s += __shfl_xor(s, off, 32);
  if (lane == 0) u[row] = s;
}

// ---------------------------------------------------------------- weight f32 -> bf16 (L2-resident)
__global__ void cvt_bf16x4_kernel(const float* __restrict__ s, __bf16* __restrict__ d, int n4) {
  const int i = blockIdx.x * blockDim.x + threadIdx.x;
  if (i >= n4) return;
  const float4 v = ((const float4*)s)[i];
  uint2 o; o.x = pack2(v.x, v.y); o.y = pack2(v.z, v.w);
  ((uint2*)d)[i] = o;
}

// ---------------------------------------------------------------- new_m = m @ AT^T + u * BT^T
// 64x128 block tile, 8 waves (2x4), wave = 32x32 via 2x2 wmma frags.
// Double-buffered LDS filled by global_load_async_to_lds_b128 (3 async ops /
// wave / K-slab: 1 for A 64x32, 2 for W 128x32).
__global__ void lmu_m_kernel(const __bf16* __restrict__ mb, const __bf16* __restrict__ ATb,
                             const float* __restrict__ u, const float* __restrict__ BT,
                             float* __restrict__ new_m, __bf16* __restrict__ nmb) {
  __shared__ __bf16 As[2][64 * LDSS];
  __shared__ __bf16 Ws[2][128 * LDSS];
  const int tid = threadIdx.x, lane = tid & 31, wave = tid >> 5;
  const int wave_m = wave >> 2, wave_n = wave & 3;
  const int m0 = blockIdx.x * 64, n0 = blockIdx.y * 128;

  const int r   = tid >> 2;              // 0..63
  const unsigned cB = (unsigned)(tid & 3) * 16u;   // 16B chunk within a 64B row

  auto issue = [&](int buf, int kk) {
    {
      const unsigned dst = lds_off32(&As[buf][0]) + (unsigned)r * (LDSS * 2) + cB;
      ASYNC_LD_B128(dst, gaddr64((const char*)(mb + (size_t)(m0 + r) * D_M + kk) + cB));
    }
#pragma unroll
    for (int hh = 0; hh < 2; ++hh) {
      const int rr = r + hh * 64;        // 0..127
      const unsigned dst = lds_off32(&Ws[buf][0]) + (unsigned)rr * (LDSS * 2) + cB;
      ASYNC_LD_B128(dst, gaddr64((const char*)(ATb + (size_t)(n0 + rr) * D_M + kk) + cB));
    }
  };

  v8f acc[2][2];
#pragma unroll
  for (int i = 0; i < 2; ++i)
#pragma unroll
    for (int j = 0; j < 2; ++j)
#pragma unroll
      for (int g = 0; g < 8; ++g) acc[i][j][g] = 0.f;

  issue(0, 0);
  int buf = 0;
  for (int kk = 0; kk < D_M; kk += 32, buf ^= 1) {
    const bool last = (kk + 32 >= D_M);
    if (!last) { issue(buf ^ 1, kk + 32); WAIT_ASYNC(0x3); }
    else       { WAIT_ASYNC(0x0); }
    __syncthreads();
    const __bf16* Ab = As[buf];
    const __bf16* Wb = Ws[buf];
    v16bf af[2], bfr[2];
    af[0]  = frag_a(Ab, wave_m * 32,      lane);
    af[1]  = frag_a(Ab, wave_m * 32 + 16, lane);
    bfr[0] = frag_b(Wb, wave_n * 32,      lane);
    bfr[1] = frag_b(Wb, wave_n * 32 + 16, lane);
#pragma unroll
    for (int i = 0; i < 2; ++i)
#pragma unroll
      for (int j = 0; j < 2; ++j)
        acc[i][j] = __builtin_amdgcn_wmma_f32_16x16x32_bf16(
            false, af[i], false, bfr[j], (short)0, acc[i][j], false, false);
    __syncthreads();
  }

  const int col = lane & 15, half = lane >> 4;
#pragma unroll
  for (int i = 0; i < 2; ++i)
#pragma unroll
    for (int j = 0; j < 2; ++j) {
      const int gm0 = m0 + wave_m * 32 + i * 16 + half * 8;
      const int gn  = n0 + wave_n * 32 + j * 16 + col;
      const float bt = BT[gn];
#pragma unroll
      for (int g = 0; g < 8; ++g) {
        const int gm = gm0 + g;
        const float v = acc[i][j][g] + u[gm] * bt;
        new_m[(size_t)gm * D_M + gn] = v;      // f32 output (tuple elem 1)
        nmb[(size_t)gm * D_M + gn]  = f2bf(v); // bf16 feed for the h-GEMM
      }
    }
}

// ---------------------------------------------------------------- new_h = sigmoid([x|h|new_m] @ [Wx|Wh|Wm]^T)
__global__ void lmu_h_kernel(const __bf16* __restrict__ xb, const __bf16* __restrict__ hb,
                             const __bf16* __restrict__ nmb,
                             const __bf16* __restrict__ Wxb, const __bf16* __restrict__ Whb,
                             const __bf16* __restrict__ Wmb, float* __restrict__ new_h) {
  __shared__ __bf16 As[2][64 * LDSS];
  __shared__ __bf16 Ws[2][128 * LDSS];
  const int tid = threadIdx.x, lane = tid & 31, wave = tid >> 5;
  const int wave_m = wave >> 2, wave_n = wave & 3;
  const int m0 = blockIdx.x * 64, n0 = blockIdx.y * 128;

  const int r   = tid >> 2;
  const unsigned cB = (unsigned)(tid & 3) * 16u;

  // Segment boundaries (128, 640) are multiples of 32: a slab never straddles.
  auto issue = [&](int buf, int kk) {
    const __bf16* Aseg; const __bf16* Wseg; int ld, ka;
    if (kk < K_X)            { Aseg = xb;  Wseg = Wxb; ld = K_X; ka = kk; }
    else if (kk < K_X + N_H) { Aseg = hb;  Wseg = Whb; ld = N_H; ka = kk - K_X; }
    else                     { Aseg = nmb; Wseg = Wmb; ld = D_M; ka = kk - K_X - N_H; }
    {
      const unsigned dst = lds_off32(&As[buf][0]) + (unsigned)r * (LDSS * 2) + cB;
      ASYNC_LD_B128(dst, gaddr64((const char*)(Aseg + (size_t)(m0 + r) * ld + ka) + cB));
    }
#pragma unroll
    for (int hh = 0; hh < 2; ++hh) {
      const int rr = r + hh * 64;
      const unsigned dst = lds_off32(&Ws[buf][0]) + (unsigned)rr * (LDSS * 2) + cB;
      ASYNC_LD_B128(dst, gaddr64((const char*)(Wseg + (size_t)(n0 + rr) * ld + ka) + cB));
    }
  };

  v8f acc[2][2];
#pragma unroll
  for (int i = 0; i < 2; ++i)
#pragma unroll
    for (int j = 0; j < 2; ++j)
#pragma unroll
      for (int g = 0; g < 8; ++g) acc[i][j][g] = 0.f;

  issue(0, 0);
  int buf = 0;
  for (int kk = 0; kk < K_TOT; kk += 32, buf ^= 1) {
    const bool last = (kk + 32 >= K_TOT);
    if (!last) { issue(buf ^ 1, kk + 32); WAIT_ASYNC(0x3); }
    else       { WAIT_ASYNC(0x0); }
    __syncthreads();
    const __bf16* Ab = As[buf];
    const __bf16* Wb = Ws[buf];
    v16bf af[2], bfr[2];
    af[0]  = frag_a(Ab, wave_m * 32,      lane);
    af[1]  = frag_a(Ab, wave_m * 32 + 16, lane);
    bfr[0] = frag_b(Wb, wave_n * 32,      lane);
    bfr[1] = frag_b(Wb, wave_n * 32 + 16, lane);
#pragma unroll
    for (int i = 0; i < 2; ++i)
#pragma unroll
      for (int j = 0; j < 2; ++j)
        acc[i][j] = __builtin_amdgcn_wmma_f32_16x16x32_bf16(
            false, af[i], false, bfr[j], (short)0, acc[i][j], false, false);
    __syncthreads();
  }

  const int col = lane & 15, half = lane >> 4;
#pragma unroll
  for (int i = 0; i < 2; ++i)
#pragma unroll
    for (int j = 0; j < 2; ++j) {
      const int gm0 = m0 + wave_m * 32 + i * 16 + half * 8;
      const int gn  = n0 + wave_n * 32 + j * 16 + col;
#pragma unroll
      for (int g = 0; g < 8; ++g) {
        const float v = acc[i][j][g];
        new_h[(size_t)(gm0 + g) * N_H + gn] = 1.f / (1.f + __expf(-v));
      }
    }
}

// ---------------------------------------------------------------- launch
extern "C" void kernel_launch(void* const* d_in, const int* in_sizes, int n_in,
                              void* d_out, int out_size, void* d_ws, size_t ws_size,
                              hipStream_t stream) {
  (void)in_sizes; (void)n_in; (void)out_size; (void)ws_size;
  const float* x  = (const float*)d_in[0];
  const float* h  = (const float*)d_in[1];
  const float* m  = (const float*)d_in[2];
  const float* Wx = (const float*)d_in[3];
  const float* Wh = (const float*)d_in[4];
  const float* Wm = (const float*)d_in[5];
  const float* ex = (const float*)d_in[6];
  const float* eh = (const float*)d_in[7];
  const float* em = (const float*)d_in[8];
  const float* AT = (const float*)d_in[9];
  const float* BT = (const float*)d_in[10];

  float* new_h = (float*)d_out;                          // (B, N) first in tuple
  float* new_m = (float*)d_out + (size_t)B_BATCH * N_H;  // (B, D) second

  // Workspace carve-up (all 256B aligned; ~37 MB total).
  char* ws = (char*)d_ws;
  size_t off = 0;
  auto take = [&](size_t bytes) { char* p = ws + off; off = (off + bytes + 255) & ~(size_t)255; return p; };
  float*  u   = (float*)take((size_t)B_BATCH * 4);
  __bf16* xb  = (__bf16*)take((size_t)B_BATCH * K_X * 2);
  __bf16* hb  = (__bf16*)take((size_t)B_BATCH * N_H * 2);
  __bf16* mb  = (__bf16*)take((size_t)B_BATCH * D_M * 2);
  __bf16* nmb = (__bf16*)take((size_t)B_BATCH * D_M * 2);
  __bf16* Wxb = (__bf16*)take((size_t)N_H * K_X * 2);
  __bf16* Whb = (__bf16*)take((size_t)N_H * N_H * 2);
  __bf16* Wmb = (__bf16*)take((size_t)N_H * D_M * 2);
  __bf16* ATb = (__bf16*)take((size_t)D_M * D_M * 2);

  // 1) fused u reduction + activation bf16 conversion (one wave32 per row).
  lmu_prep_kernel<<<B_BATCH / 8, 256, 0, stream>>>(x, h, m, ex, eh, em, xb, hb, mb, u);

  // 2) weight conversions (tiny, L2-resident).
  cvt_bf16x4_kernel<<<(N_H * K_X / 4) / 256, 256, 0, stream>>>(Wx, Wxb, N_H * K_X / 4);
  cvt_bf16x4_kernel<<<(N_H * N_H / 4) / 256, 256, 0, stream>>>(Wh, Whb, N_H * N_H / 4);
  cvt_bf16x4_kernel<<<(N_H * D_M / 4) / 256, 256, 0, stream>>>(Wm, Wmb, N_H * D_M / 4);
  cvt_bf16x4_kernel<<<(D_M * D_M / 4) / 256, 256, 0, stream>>>(AT, ATb, D_M * D_M / 4);

  // 3) new_m GEMM (+u*BT epilogue, dual f32/bf16 store): 256 x 2 blocks.
  lmu_m_kernel<<<dim3(B_BATCH / 64, D_M / 128), 256, 0, stream>>>(mb, ATb, u, BT, new_m, nmb);

  // 4) new_h GEMM (K=896 concat) + sigmoid: 256 x 4 blocks.
  lmu_h_kernel<<<dim3(B_BATCH / 64, N_H / 128), 256, 0, stream>>>(xb, hb, nmb, Wxb, Whb, Wmb, new_h);
}
